// Head_49727131353090
// MI455X (gfx1250) — matrix operand order repaired and checked by reading, not compiled, not run on hardware
//
#include <hip/hip_runtime.h>

#define EMBED 1024
#define HEAD  64
#define SEQ   2048
#define BATCH 8

typedef __attribute__((ext_vector_type(16))) __bf16        v16bf;
typedef __attribute__((ext_vector_type(2)))  __bf16        v2bf;
typedef __attribute__((ext_vector_type(8)))  float         v8f;
typedef __attribute__((ext_vector_type(4)))  unsigned int  u32x4;
typedef __attribute__((ext_vector_type(4)))  float         f32x4;

union ABReg {
    u32x4        q[2];
    unsigned int u[8];
    v16bf        v;
};

// Native bf16 conversion: lowers to v_cvt_pk_bf16_f32 on gfx1250 (verified).
__device__ __forceinline__ unsigned short f2bf(float a) {
    __bf16 h = (__bf16)a;
    return __builtin_bit_cast(unsigned short, h);
}
__device__ __forceinline__ unsigned int pack_bf16(float a, float b) {
    v2bf t;
    t[0] = (__bf16)a;
    t[1] = (__bf16)b;
    return __builtin_bit_cast(unsigned int, t);
}

__device__ __forceinline__ void wave_fence() {
    asm volatile("" ::: "memory");
    __builtin_amdgcn_wave_barrier();
    asm volatile("" ::: "memory");
}
// Scheduling-region boundary (no hardware wait): stop the machine scheduler
// from sinking prefetch loads past this point.
__device__ __forceinline__ void sched_fence() {
    __builtin_amdgcn_sched_barrier(0);
}

// ---------------------------------------------------------------------------
// Kernel 1: transpose + convert weights to bf16: Wt[m][h][e], m: 0=Q,1=K,2=V
// ---------------------------------------------------------------------------
__global__ void wt_kernel(const float* __restrict__ Wq, const float* __restrict__ Wk,
                          const float* __restrict__ Wv, unsigned short* __restrict__ Wt) {
    int idx = blockIdx.x * blockDim.x + threadIdx.x;
    if (idx >= 3 * HEAD * EMBED) return;
    int m   = idx / (HEAD * EMBED);
    int rem = idx - m * (HEAD * EMBED);
    int h   = rem / EMBED;
    int e   = rem - h * EMBED;
    const float* W = (m == 0) ? Wq : ((m == 1) ? Wk : Wv);
    Wt[idx] = f2bf(W[e * HEAD + h]);
}

// ---------------------------------------------------------------------------
// Kernel 2: fused projection, one matrix per wave, software-pipelined.
// Block = 192 threads (6 waves): wave w -> matrix w%3, row-tile w/3.
// Block covers 32 rows; grid = (B*T)/32 = 512 blocks.
// ---------------------------------------------------------------------------
__device__ __forceinline__ void loadBtiles(ABReg dst[4], const unsigned short* wbase, int e) {
    #pragma unroll
    for (int ht = 0; ht < 4; ++ht) {
        const unsigned short* wp = wbase + (size_t)ht * 16 * EMBED + e;
        dst[ht].q[0] = *(const u32x4*)wp;
        dst[ht].q[1] = *(const u32x4*)(wp + 16);
    }
}
__device__ __forceinline__ void loadAfloats(f32x4 dst[4], const float* xrow, int e) {
    const float* xr = xrow + e;
    dst[0] = ((const f32x4*)xr)[0];
    dst[1] = ((const f32x4*)xr)[1];
    dst[2] = ((const f32x4*)(xr + 16))[0];
    dst[3] = ((const f32x4*)(xr + 16))[1];
}
__device__ __forceinline__ ABReg packA(const f32x4 af[4]) {
    ABReg a;
    a.u[0] = pack_bf16(af[0][0], af[0][1]); a.u[1] = pack_bf16(af[0][2], af[0][3]);
    a.u[2] = pack_bf16(af[1][0], af[1][1]); a.u[3] = pack_bf16(af[1][2], af[1][3]);
    a.u[4] = pack_bf16(af[2][0], af[2][1]); a.u[5] = pack_bf16(af[2][2], af[2][3]);
    a.u[6] = pack_bf16(af[3][0], af[3][1]); a.u[7] = pack_bf16(af[3][2], af[3][3]);
    return a;
}

__global__ __launch_bounds__(192) void proj_kernel(
    const float* __restrict__ x, const unsigned short* __restrict__ Wt,
    unsigned short* __restrict__ Qbf, unsigned short* __restrict__ Kbf,
    unsigned short* __restrict__ Vt)
{
    const int lane = threadIdx.x & 31;
    const int wave = threadIdx.x >> 5;
    const int half = lane >> 4;
    const int ln   = lane & 15;
    const int mat  = wave % 3;                          // 0=Q, 1=K, 2=V
    const int m0   = blockIdx.x * 32 + (wave / 3) * 16; // global row base

    const float*          xrow  = x + (size_t)(m0 + ln) * EMBED + half * 8;
    const unsigned short* wbase = Wt + (size_t)(mat * HEAD + ln) * EMBED + half * 8;

    v8f acc[4];
    #pragma unroll
    for (int ht = 0; ht < 4; ++ht) acc[ht] = v8f{};

    f32x4 afA[4], afB[4];
    ABReg bbA[4], bbB[4];
    loadAfloats(afA, xrow, 0);
    loadBtiles(bbA, wbase, 0);

    for (int e = 0; e < EMBED; e += 64) {
        // half 1: prefetch e+32, then consume (afA, bbA)
        loadAfloats(afB, xrow, e + 32);
        loadBtiles(bbB, wbase, e + 32);
        sched_fence();                      // keep prefetch issued before compute
        {
            ABReg a = packA(afA);
            #pragma unroll
            for (int ht = 0; ht < 4; ++ht)
                acc[ht] = __builtin_amdgcn_wmma_f32_16x16x32_bf16(
                    false, a.v, false, bbA[ht].v, (short)0, acc[ht], false, false);
        }
        sched_fence();
        // half 2: prefetch e+64 (wrap -> harmless reload of 0), consume (afB, bbB)
        const int e2 = (e + 64 < EMBED) ? e + 64 : 0;
        loadAfloats(afA, xrow, e2);
        loadBtiles(bbA, wbase, e2);
        sched_fence();
        {
            ABReg a = packA(afB);
            #pragma unroll
            for (int ht = 0; ht < 4; ++ht)
                acc[ht] = __builtin_amdgcn_wmma_f32_16x16x32_bf16(
                    false, a.v, false, bbB[ht].v, (short)0, acc[ht], false, false);
        }
        sched_fence();
    }

    if (mat < 2) {
        unsigned short* dst = (mat == 0) ? Qbf : Kbf;
        #pragma unroll
        for (int ht = 0; ht < 4; ++ht) {
            const int h = ht * 16 + ln;
            #pragma unroll
            for (int r = 0; r < 8; ++r) {
                const int row = m0 + r + half * 8;
                dst[(size_t)row * HEAD + h] = f2bf(acc[ht][r]);
            }
        }
    } else {
        // V transposed: lane = fixed h, 8 consecutive t -> one b128 store per tile
        const int b  = m0 / SEQ;
        const int t0 = (m0 % SEQ) + half * 8;
        #pragma unroll
        for (int ht = 0; ht < 4; ++ht) {
            const int h = ht * 16 + ln;
            u32x4 pk;
            pk[0] = pack_bf16(acc[ht][0], acc[ht][1]);
            pk[1] = pack_bf16(acc[ht][2], acc[ht][3]);
            pk[2] = pack_bf16(acc[ht][4], acc[ht][5]);
            pk[3] = pack_bf16(acc[ht][6], acc[ht][7]);
            *(u32x4*)(Vt + (size_t)(b * HEAD + h) * SEQ + t0) = pk;
        }
    }
}

// ---------------------------------------------------------------------------
// Kernel 3: causal flash attention.  Block = (qb, b), 4 independent waves,
// wave owns 16 queries.  Batched B-tile loads, Bv prefetched behind softmax,
// wave-local LDS fence (no __syncthreads).
// ---------------------------------------------------------------------------
__global__ __launch_bounds__(128) void attn_kernel(
    const unsigned short* __restrict__ Qbf, const unsigned short* __restrict__ Kbf,
    const unsigned short* __restrict__ Vt, float* __restrict__ out)
{
    __shared__ __align__(16) unsigned short p_lds[4][16][64];

    const int lane = threadIdx.x & 31;
    const int wave = threadIdx.x >> 5;
    const int half = lane >> 4;
    const int ln   = lane & 15;
    const int qb   = blockIdx.x;
    const int b    = blockIdx.y;
    const int q0   = qb * 64 + wave * 16;

    // Q A-operands (persist in registers)
    ABReg Aq[2];
    #pragma unroll
    for (int c = 0; c < 2; ++c) {
        const unsigned short* qp =
            Qbf + (size_t)(b * SEQ + q0 + ln) * HEAD + c * 32 + half * 8;
        Aq[c].q[0] = *(const u32x4*)qp;
        Aq[c].q[1] = *(const u32x4*)(qp + 16);
    }

    v8f O[4];
    float mrow[8], lrow[8];
    #pragma unroll
    for (int ht = 0; ht < 4; ++ht) O[ht] = v8f{};
    #pragma unroll
    for (int r = 0; r < 8; ++r) { mrow[r] = -__builtin_inff(); lrow[r] = 0.0f; }

    const float scale = 0.03125f;   // EMBED^-0.5

    for (int kb = 0; kb <= qb; ++kb) {
        const int kbase = kb * 64;

        // ---- batched Bk loads (one clause), then 8 S-WMMAs ----
        ABReg Bk[8];
        #pragma unroll
        for (int kt = 0; kt < 4; ++kt)
            #pragma unroll
            for (int c = 0; c < 2; ++c) {
                const unsigned short* kp =
                    Kbf + (size_t)(b * SEQ + kbase + kt * 16 + ln) * HEAD + c * 32 + half * 8;
                Bk[kt * 2 + c].q[0] = *(const u32x4*)kp;
                Bk[kt * 2 + c].q[1] = *(const u32x4*)(kp + 16);
            }
        sched_fence();
        v8f S[4];
        #pragma unroll
        for (int kt = 0; kt < 4; ++kt) {
            v8f s = v8f{};
            s = __builtin_amdgcn_wmma_f32_16x16x32_bf16(
                false, Aq[0].v, false, Bk[kt * 2 + 0].v, (short)0, s, false, false);
            s = __builtin_amdgcn_wmma_f32_16x16x32_bf16(
                false, Aq[1].v, false, Bk[kt * 2 + 1].v, (short)0, s, false, false);
            S[kt] = s;
        }

        // ---- prefetch Bv now; softmax VALU hides the latency ----
        ABReg Bv[8];
        #pragma unroll
        for (int c2 = 0; c2 < 2; ++c2)
            #pragma unroll
            for (int ht = 0; ht < 4; ++ht) {
                const unsigned short* vp =
                    Vt + (size_t)(b * HEAD + ht * 16 + ln) * SEQ + kbase + c2 * 32 + half * 8;
                Bv[c2 * 4 + ht].q[0] = *(const u32x4*)vp;
                Bv[c2 * 4 + ht].q[1] = *(const u32x4*)(vp + 16);
            }
        sched_fence();                      // Bv issue stays ahead of softmax

        // ---- scale + causal mask ----
        if (kb == qb) {
            #pragma unroll
            for (int kt = 0; kt < 4; ++kt)
                #pragma unroll
                for (int r = 0; r < 8; ++r) {
                    const int key = kbase + kt * 16 + ln;
                    const int qq  = q0 + r + half * 8;
                    const float s = S[kt][r] * scale;
                    S[kt][r] = (key <= qq) ? s : -__builtin_inff();
                }
        } else {
            #pragma unroll
            for (int kt = 0; kt < 4; ++kt)
                #pragma unroll
                for (int r = 0; r < 8; ++r)
                    S[kt][r] *= scale;
        }

        // ---- online softmax ----
        float mnew[8], alpha[8];
        #pragma unroll
        for (int r = 0; r < 8; ++r) {
            float v = fmaxf(fmaxf(S[0][r], S[1][r]), fmaxf(S[2][r], S[3][r]));
            #pragma unroll
            for (int off = 1; off < 16; off <<= 1)
                v = fmaxf(v, __shfl_xor(v, off, 32));
            mnew[r]  = fmaxf(mrow[r], v);
            alpha[r] = __expf(mrow[r] - mnew[r]);
            mrow[r]  = mnew[r];
        }
        #pragma unroll
        for (int kt = 0; kt < 4; ++kt)
            #pragma unroll
            for (int r = 0; r < 8; ++r)
                S[kt][r] = __expf(S[kt][r] - mnew[r]);
        #pragma unroll
        for (int r = 0; r < 8; ++r) {
            float ps = (S[0][r] + S[1][r]) + (S[2][r] + S[3][r]);
            #pragma unroll
            for (int off = 1; off < 16; off <<= 1)
                ps += __shfl_xor(ps, off, 32);
            lrow[r] = lrow[r] * alpha[r] + ps;
        }
        #pragma unroll
        for (int ht = 0; ht < 4; ++ht)
            #pragma unroll
            for (int r = 0; r < 8; ++r)
                O[ht][r] *= alpha[r];

        // ---- P: C-layout -> A-layout via wave-private LDS tile ----
        #pragma unroll
        for (int kt = 0; kt < 4; ++kt)
            #pragma unroll
            for (int r = 0; r < 8; ++r)
                p_lds[wave][r + half * 8][kt * 16 + ln] = f2bf(S[kt][r]);
        wave_fence();   // DS is in-order per wave; fence stops compiler reordering

        // ---- O += P V ----
        #pragma unroll
        for (int c2 = 0; c2 < 2; ++c2) {
            const unsigned short* pp = &p_lds[wave][ln][c2 * 32 + half * 8];
            ABReg Ap;
            Ap.q[0] = *(const u32x4*)pp;
            Ap.q[1] = *(const u32x4*)(pp + 16);
            #pragma unroll
            for (int ht = 0; ht < 4; ++ht)
                O[ht] = __builtin_amdgcn_wmma_f32_16x16x32_bf16(
                    false, Ap.v, false, Bv[c2 * 4 + ht].v, (short)0, O[ht], false, false);
        }
        wave_fence();   // keep next iteration's DS stores behind these loads
    }

    // ---- epilogue ----
    #pragma unroll
    for (int r = 0; r < 8; ++r) {
        const float inv = 1.0f / lrow[r];
        const int   row = b * SEQ + q0 + r + half * 8;
        #pragma unroll
        for (int ht = 0; ht < 4; ++ht)
            out[(size_t)row * HEAD + ht * 16 + ln] = O[ht][r] * inv;
    }
}

// ---------------------------------------------------------------------------
extern "C" void kernel_launch(void* const* d_in, const int* in_sizes, int n_in,
                              void* d_out, int out_size, void* d_ws, size_t ws_size,
                              hipStream_t stream) {
    const float* x  = (const float*)d_in[0];
    const float* Wk = (const float*)d_in[1];
    const float* Wq = (const float*)d_in[2];
    const float* Wv = (const float*)d_in[3];

    unsigned short* ws  = (unsigned short*)d_ws;
    unsigned short* Wt  = ws;                                   // 3*64*1024
    unsigned short* Qbf = Wt + (size_t)3 * HEAD * EMBED;        // B*T*H
    unsigned short* Kbf = Qbf + (size_t)BATCH * SEQ * HEAD;
    unsigned short* Vt  = Kbf + (size_t)BATCH * SEQ * HEAD;     // [B][H][T]

    wt_kernel<<<(3 * HEAD * EMBED + 255) / 256, 256, 0, stream>>>(Wq, Wk, Wv, Wt);
    proj_kernel<<<(BATCH * SEQ) / 32, 192, 0, stream>>>(x, Wt, Qbf, Kbf, Vt);
    attn_kernel<<<dim3(SEQ / 64, BATCH), 128, 0, stream>>>(Qbf, Kbf, Vt, (float*)d_out);
}
